// RGB3D_17652315586965
// MI455X (gfx1250) — compile-verified, matrix operand order checked by reading, not run on hardware
//
#include <hip/hip_runtime.h>
#include <hip/hip_bf16.h>

typedef __attribute__((ext_vector_type(8))) int v8i;

#define NBINS   17
#define NSAMP   625
#define IMG_H   512
#define IMG_W   512

// One block per (batch, channel). 64 threads = 2 wave32.
// Each wave handles 5 chunks of 64 samples; histogram counting is done as
// D = A x 1 with V_WMMA_I32_16X16X64_IU8 where A[m][k] = (bin_k == m).
__global__ __launch_bounds__(64) void RGB3D_hist_wmma_kernel(
    const float* __restrict__ imgs,
    const int*   __restrict__ sample_x,
    const int*   __restrict__ sample_y,
    float*       __restrict__ out)
{
    const int b    = blockIdx.x;        // batch index
    const int c    = blockIdx.y;        // channel 0..2
    const int tid  = threadIdx.x;       // 0..63
    const int wave = tid >> 5;          // 0..1
    const int lane = tid & 31;

    __shared__ unsigned int  hist[NBINS];
    __shared__ unsigned char bins[2][64];   // per-wave chunk bins (bytes)

    if (tid < NBINS) hist[tid] = 0u;

    const int* __restrict__ sx = sample_x + (size_t)b * NSAMP;
    const int* __restrict__ sy = sample_y + (size_t)b * NSAMP;
    __builtin_prefetch(sx, 0, 0);           // global_prefetch_b8
    __builtin_prefetch(sy, 0, 0);

    __syncthreads();

    v8i acc = {};                           // C/D accumulator (i32 16x16)
    v8i bones;
    #pragma unroll
    for (int j = 0; j < 8; ++j) bones[j] = 0x01010101;  // B = all ones

    const unsigned int m  = (unsigned int)(lane & 15);  // my A-matrix row
    const int          g2 = (lane >> 4) << 1;           // +2 dwords for lanes 16..31
    unsigned int c16 = 0;                               // bin==16 counter (exactness)

    // 10 chunks of 64 samples; wave w takes chunks {2i + w}, i = 0..4.
    #pragma unroll 1
    for (int i = 0; i < 5; ++i) {
        const int chunk = 2 * i + wave;
        const int s0 = chunk * 64 + 2 * lane;
        const int s1 = s0 + 1;
        // clamp to stay in-bounds; invalid lanes are neutralized via bin=0xFF
        const int t0 = (s0 < NSAMP - 1) ? s0 : (NSAMP - 1);
        const int t1 = (s1 < NSAMP - 1) ? s1 : (NSAMP - 1);
        const int x0 = sx[t0], y0 = sy[t0];
        const int x1 = sx[t1], y1 = sy[t1];
        const float p0 = imgs[(((size_t)b * IMG_H + (size_t)x0) * IMG_W + (size_t)y0) * 3 + c];
        const float p1 = imgs[(((size_t)b * IMG_H + (size_t)x1) * IMG_W + (size_t)y1) * 3 + c];
        const int bin0 = (int)floorf(p0 * 0.0625f);
        const int bin1 = (int)floorf(p1 * 0.0625f);
        // one_hot semantics: bins outside [0,16] contribute nothing
        const unsigned int u0 =
            (s0 < NSAMP && bin0 >= 0 && bin0 <= 16) ? (unsigned int)bin0 : 0xFFu;
        const unsigned int u1 =
            (s1 < NSAMP && bin1 >= 0 && bin1 <= 16) ? (unsigned int)bin1 : 0xFFu;
        c16 += (u0 == 16u) + (u1 == 16u);   // WMMA rows cover only m=0..15

        // publish the 64 chunk bins: byte K of the chunk = bin of sample K
        ((unsigned short*)bins[wave])[lane] = (unsigned short)(u0 | (u1 << 8));
        __syncthreads();

        // Build A (8-bit 16x64, ISA 7.12.2): lanes 0-15 row m=lane hold
        // K-dwords {0,1,4,5,8,9,12,13}; lanes 16-31 row m=lane-16 hold +2.
        const unsigned int* bw = (const unsigned int*)bins[wave];
        const int dbase[8] = {0, 1, 4, 5, 8, 9, 12, 13};
        v8i A;
        #pragma unroll
        for (int j = 0; j < 8; ++j) {
            const unsigned int w = bw[dbase[j] + g2];
            unsigned int a = 0u;
            a |= (( w        & 0xFFu) == m) ? 0x00000001u : 0u;
            a |= (((w >> 8)  & 0xFFu) == m) ? 0x00000100u : 0u;
            a |= (((w >> 16) & 0xFFu) == m) ? 0x00010000u : 0u;
            a |= (( w >> 24)          == m) ? 0x01000000u : 0u;
            A[j] = (int)a;
        }

        // D[m][n] = sum_k A[m][k] * 1  -> count of bin m, replicated over n.
        // (sgn_a, A, sgn_b, B, C, reuse_a, reuse_b) ; unsigned x unsigned.
        acc = __builtin_amdgcn_wmma_i32_16x16x64_iu8(false, A, false, bones,
                                                     acc, false, false);
        __syncthreads();
    }

    // C/D layout: VGPR r -> lanes 0-15: D[r][lane]; lanes 16-31: D[8+r][lane-16].
    // Column 0 => lane 0 holds bins 0..7, lane 16 holds bins 8..15.
    if (lane == 0) {
        #pragma unroll
        for (int r = 0; r < 8; ++r) atomicAdd(&hist[r], (unsigned int)acc[r]);
    }
    if (lane == 16) {
        #pragma unroll
        for (int r = 0; r < 8; ++r) atomicAdd(&hist[8 + r], (unsigned int)acc[r]);
    }
    if (c16) atomicAdd(&hist[16], c16);
    __syncthreads();

    if (tid < NBINS)
        out[(size_t)b * (3 * NBINS) + c * NBINS + tid] = (float)hist[tid];
}

extern "C" void kernel_launch(void* const* d_in, const int* in_sizes, int n_in,
                              void* d_out, int out_size, void* d_ws, size_t ws_size,
                              hipStream_t stream) {
    const float* imgs = (const float*)d_in[0];
    const int*   sx   = (const int*)d_in[1];
    const int*   sy   = (const int*)d_in[2];
    float*       out  = (float*)d_out;

    const int B = in_sizes[1] / NSAMP;   // 128
    dim3 grid((unsigned)B, 3u, 1u);
    RGB3D_hist_wmma_kernel<<<grid, 64, 0, stream>>>(imgs, sx, sy, out);
}
